// DmtlnModel_49727131353684
// MI455X (gfx1250) — matrix operand order repaired and checked by reading, not compiled, not run on hardware
//
#include <hip/hip_runtime.h>

// ---- types for WMMA fragments -------------------------------------------
typedef __attribute__((ext_vector_type(16))) _Float16 v16h;
typedef __attribute__((ext_vector_type(8)))  _Float16 v8h;
typedef __attribute__((ext_vector_type(8)))  float    v8f;
typedef __attribute__((ext_vector_type(4)))  float    v4f;

#define TM    64      // samples per workgroup
#define F_DIM 16      // features (8 cat + 8 cont)
#define D_DIM 32      // embedding dim
#define K1    512     // F*D
#define H1D   64
#define H2D   32
#define VOC   1000

struct Phase2 {
  _Float16 h1[TM][H1D];     // 8 KB
  _Float16 h2[TM][H2D];     // 4 KB
  float    base[TM][D_DIM]; // 8 KB (FM + linear terms)
};
union SharedU {
  _Float16 in[TM][K1];      // 64 KB, dead after GEMM1
  Phase2   p2;
};

// B fragment (K=32 x N=16, f16) built from 16 contiguous f32 of a weight row.
// Lane (n = lane&15, hi = lane>>4) holds B[k0 + hi*16 + i][n], i = 0..15.
__device__ __forceinline__ v16h load_b16_from_f32(const float* __restrict__ p) {
  union { v16h v; _Float16 e[16]; } u;
#pragma unroll
  for (int c = 0; c < 4; ++c) {
    v4f f = ((const v4f*)p)[c];
    u.e[4*c+0] = (_Float16)f[0];
    u.e[4*c+1] = (_Float16)f[1];
    u.e[4*c+2] = (_Float16)f[2];
    u.e[4*c+3] = (_Float16)f[3];
  }
  return u.v;
}

// A fragment (M=16 x K=32, f16): lane hi-half 0 holds
// K {k0..k0+15}, hi-half 1 holds K {k0+16..k0+31} (matches B fragment).
__device__ __forceinline__ v16h load_a_h16(const _Float16* row, int k0, int hi) {
  union { v16h v; v8h h[2]; } u;
  const _Float16* p = row + k0 + hi * 16;
  u.h[0] = *(const v8h*)(p);
  u.h[1] = *(const v8h*)(p + 8);
  return u.v;
}

__device__ __forceinline__ v8f wmma_f16(v16h a, v16h b, v8f c) {
  return __builtin_amdgcn_wmma_f32_16x16x32_f16(false, a, false, b,
                                                (short)0, c, false, false);
}

__global__ __launch_bounds__(256)
void dmtln_fused(const float* __restrict__ x,
                 const float* __restrict__ cat_emb,
                 const float* __restrict__ cat_lin,
                 const float* __restrict__ cont_w,
                 const float* __restrict__ cont_b,
                 const float* __restrict__ cont_lin_w,
                 const float* __restrict__ cont_lin_b,
                 const float* __restrict__ fin_bias,
                 const float* __restrict__ dnn1_w,
                 const float* __restrict__ dnn1_b,
                 const float* __restrict__ bn1_g, const float* __restrict__ bn1_bt,
                 const float* __restrict__ bn1_m, const float* __restrict__ bn1_v,
                 const float* __restrict__ dnn2_w,
                 const float* __restrict__ dnn2_b,
                 const float* __restrict__ bn2_g, const float* __restrict__ bn2_bt,
                 const float* __restrict__ bn2_m, const float* __restrict__ bn2_v,
                 const float* __restrict__ out_w,
                 const float* __restrict__ out_b,
                 float* __restrict__ out)
{
  __shared__ SharedU sh;
  const int t  = threadIdx.x;
  const int sA = t >> 2;          // sample within tile (0..63)
  const int dA = (t & 3) * 8;     // 8-wide D slice
  float baseReg[8];               // FM + linear, survives until stashed

  // ---------------- Phase A: embeddings, FM, linear terms ----------------
  {
    const int g = blockIdx.x * TM + sA;                  // global sample (n*L + l)
    const float* xr = x + (size_t)g * F_DIM;
    float ssum[8], ssq[8], lin[8];
#pragma unroll
    for (int i = 0; i < 8; ++i) { ssum[i] = 0.f; ssq[i] = 0.f; lin[i] = 0.f; }

#pragma unroll
    for (int f = 0; f < 8; ++f) {                        // categorical
      const int idx = (int)xr[f];
      const float* ep = cat_emb + ((size_t)(f * VOC + idx)) * D_DIM + dA;
      const float* lp = cat_lin + ((size_t)(f * VOC + idx)) * D_DIM + dA;
      v8h ev;
#pragma unroll
      for (int i = 0; i < 8; ++i) {
        float e = ep[i];
        ssum[i] += e; ssq[i] += e * e; lin[i] += lp[i];
        ev[i] = (_Float16)e;
      }
      *(v8h*)&sh.in[sA][f * D_DIM + dA] = ev;
    }
#pragma unroll
    for (int j = 0; j < 8; ++j) {                        // continuous
      const float c   = xr[8 + j];
      const float* wp  = cont_w     + j * D_DIM + dA;
      const float* bp  = cont_b     + j * D_DIM + dA;
      const float* lwp = cont_lin_w + j * D_DIM + dA;
      const float* lbp = cont_lin_b + j * D_DIM + dA;
      v8h ev;
#pragma unroll
      for (int i = 0; i < 8; ++i) {
        float e  = fmaf(c, wp[i],  bp[i]);
        float lv = fmaf(c, lwp[i], lbp[i]);
        ssum[i] += e; ssq[i] += e * e; lin[i] += lv;
        ev[i] = (_Float16)e;
      }
      *(v8h*)&sh.in[sA][(8 + j) * D_DIM + dA] = ev;
    }
#pragma unroll
    for (int i = 0; i < 8; ++i)
      baseReg[i] = lin[i] + fin_bias[dA + i]
                 + 0.5f * (ssum[i] * ssum[i] - ssq[i]);
  }
  __syncthreads();

  const int lane = t & 31, wv = t >> 5;
  const int ln = lane & 15, hi = lane >> 4;

  // ---------------- GEMM1: (64x512) @ dnn1_w^T -> 64x64 ------------------
  const int n1 = wv & 3;               // N tile
  const int m1 = (wv >> 2) * 2;        // two M tiles per wave
  v8f acc0 = {}, acc1 = {};
  {
    const _Float16* arow0 = &sh.in[m1 * 16 + ln][0];
    const _Float16* arow1 = &sh.in[(m1 + 1) * 16 + ln][0];
    const float* brow = dnn1_w + (size_t)(n1 * 16 + ln) * K1 + hi * 16;
#pragma unroll 4
    for (int kb = 0; kb < 16; ++kb) {
      const int k0 = kb * 32;
      v16h b  = load_b16_from_f32(brow + k0);
      v16h a0 = load_a_h16(arow0, k0, hi);
      v16h a1 = load_a_h16(arow1, k0, hi);
      acc0 = wmma_f16(a0, b, acc0);
      acc1 = wmma_f16(a1, b, acc1);
    }
  }
  __syncthreads();   // all reads of sh.in done; union regions now writable

  {  // epilogue 1: bias + BN + ReLU -> f16 h1 tile
    const int o   = n1 * 16 + ln;
    const float sc  = bn1_g[o] * rsqrtf(bn1_v[o] + 1e-5f);
    const float off = bn1_bt[o] + (dnn1_b[o] - bn1_m[o]) * sc;
#pragma unroll
    for (int r = 0; r < 8; ++r) {
      const int mr = hi * 8 + r;
      sh.p2.h1[m1 * 16 + mr][o]       = (_Float16)fmaxf(fmaf(acc0[r], sc, off), 0.f);
      sh.p2.h1[(m1 + 1) * 16 + mr][o] = (_Float16)fmaxf(fmaf(acc1[r], sc, off), 0.f);
    }
  }
#pragma unroll
  for (int i = 0; i < 8; ++i) sh.p2.base[sA][dA + i] = baseReg[i];
  __syncthreads();

  // ---------------- GEMM2: (64x64) @ dnn2_w^T -> 64x32 -------------------
  const int n2 = wv & 1, m2 = wv >> 1;
  v8f acc2 = {};
  {
    const _Float16* arow = &sh.p2.h1[m2 * 16 + ln][0];
    const float* brow = dnn2_w + (size_t)(n2 * 16 + ln) * H1D + hi * 16;
#pragma unroll
    for (int kb = 0; kb < 2; ++kb) {
      v16h b = load_b16_from_f32(brow + kb * 32);
      v16h a = load_a_h16(arow, kb * 32, hi);
      acc2 = wmma_f16(a, b, acc2);
    }
  }
  {  // epilogue 2 (h2 region disjoint from h1: no barrier needed before)
    const int o   = n2 * 16 + ln;
    const float sc  = bn2_g[o] * rsqrtf(bn2_v[o] + 1e-5f);
    const float off = bn2_bt[o] + (dnn2_b[o] - bn2_m[o]) * sc;
#pragma unroll
    for (int r = 0; r < 8; ++r)
      sh.p2.h2[m2 * 16 + hi * 8 + r][o] =
          (_Float16)fmaxf(fmaf(acc2[r], sc, off), 0.f);
  }
  __syncthreads();

  // ---------------- GEMM3: (64x32) @ out_w^T -> 64x32, + base ------------
  v8f acc3 = {};
  {
    v16h b = load_b16_from_f32(out_w + (size_t)(n2 * 16 + ln) * H2D + hi * 16);
    v16h a = load_a_h16(&sh.p2.h2[m2 * 16 + ln][0], 0, hi);
    acc3 = wmma_f16(a, b, acc3);
  }
  {
    const int o = n2 * 16 + ln;
    const float ob = out_b[o];
#pragma unroll
    for (int r = 0; r < 8; ++r) {
      const int m = m2 * 16 + hi * 8 + r;
      const int g = blockIdx.x * TM + m;
      out[(size_t)g * D_DIM + o] = acc3[r] + ob + sh.p2.base[m][o];
    }
  }
}

extern "C" void kernel_launch(void* const* d_in, const int* in_sizes, int n_in,
                              void* d_out, int out_size, void* d_ws, size_t ws_size,
                              hipStream_t stream) {
  (void)n_in; (void)out_size; (void)d_ws; (void)ws_size;
  const float* x          = (const float*)d_in[0];
  const float* cat_emb    = (const float*)d_in[1];
  const float* cat_lin    = (const float*)d_in[2];
  const float* cont_w     = (const float*)d_in[3];
  const float* cont_b     = (const float*)d_in[4];
  const float* cont_lin_w = (const float*)d_in[5];
  const float* cont_lin_b = (const float*)d_in[6];
  const float* fin_bias   = (const float*)d_in[7];
  const float* dnn1_w     = (const float*)d_in[8];
  const float* dnn1_b     = (const float*)d_in[9];
  const float* bn1_g      = (const float*)d_in[10];
  const float* bn1_bt     = (const float*)d_in[11];
  const float* bn1_m      = (const float*)d_in[12];
  const float* bn1_v      = (const float*)d_in[13];
  const float* dnn2_w     = (const float*)d_in[14];
  const float* dnn2_b     = (const float*)d_in[15];
  const float* bn2_g      = (const float*)d_in[16];
  const float* bn2_bt     = (const float*)d_in[17];
  const float* bn2_m      = (const float*)d_in[18];
  const float* bn2_v      = (const float*)d_in[19];
  const float* out_w      = (const float*)d_in[20];
  const float* out_b      = (const float*)d_in[21];
  float* out = (float*)d_out;

  const int nSamples = in_sizes[0] / F_DIM;   // N*L = 196608
  const int grid = nSamples / TM;             // 3072 (exact)
  dmtln_fused<<<grid, 256, 0, stream>>>(
      x, cat_emb, cat_lin, cont_w, cont_b, cont_lin_w, cont_lin_b, fin_bias,
      dnn1_w, dnn1_b, bn1_g, bn1_bt, bn1_m, bn1_v,
      dnn2_w, dnn2_b, bn2_g, bn2_bt, bn2_m, bn2_v,
      out_w, out_b, out);
}
